// GraphTransformerLayer_35218731827282
// MI455X (gfx1250) — matrix operand order repaired and checked by reading, not compiled, not run on hardware
//
#include <hip/hip_runtime.h>
#include <hip/hip_bf16.h>

// ---------------------------------------------------------------------------
// GraphTransformerLayer for MI455X (gfx1250, wave32, WMMA).
// All GEMMs run on v_wmma_f32_16x16x32_f16 (f16 in, f32 accumulate).
// Attention is flash-style (online softmax) with K/V tiles double-buffered in
// LDS via global_load_async_to_lds_b128 (ASYNCcnt pipelining), shared by the
// 4 waves of a workgroup.
// ---------------------------------------------------------------------------

typedef _Float16 half_t;
typedef __attribute__((ext_vector_type(16))) _Float16 v16h;
typedef __attribute__((ext_vector_type(8)))  _Float16 v8h;
typedef __attribute__((ext_vector_type(8)))  float    v8f;
typedef __attribute__((ext_vector_type(4)))  float    v4f;

#define BB     2
#define NN     2048
#define DIMX   256
#define HEADS  8
#define HD     32
#define FF     1024
#define MROWS  (BB * NN)          // 4096 token rows
#define EPSV   1e-5f
#define SCALEV 0.17677669529663687f   // HEAD_DIM^-0.5

// ----- fragment loaders (ISA 7.12.2 layouts) -------------------------------
// A 16x32 f16: lane m=lane&15, K chunks [8h,8h+8) and [16+8h,16+8h+8)
__device__ __forceinline__ v16h load_frag_a(const half_t* tile, int ld, int lane) {
  const int m = lane & 15, hf = lane >> 4;
  const half_t* p = tile + (size_t)m * ld + 8 * hf;
  v8h lo = *(const v8h*)(p);
  v8h hi = *(const v8h*)(p + 16);
  return __builtin_shufflevector(lo, hi, 0,1,2,3,4,5,6,7,8,9,10,11,12,13,14,15);
}
// B 32x16 f16 from Bt row-major [Ncols][K]: lane n=lane&15, K chunks [8h,8h+8) and [16+8h,16+8h+8)
__device__ __forceinline__ v16h load_frag_b(const half_t* tile, int ld, int lane) {
  const int n = lane & 15, hf = lane >> 4;
  const half_t* p = tile + (size_t)n * ld + 8 * hf;
  v8h lo = *(const v8h*)(p);
  v8h hi = *(const v8h*)(p + 16);
  return __builtin_shufflevector(lo, hi, 0,1,2,3,4,5,6,7,8,9,10,11,12,13,14,15);
}
__device__ __forceinline__ v8f wmma16(v16h a, v16h b, v8f c) {
  return __builtin_amdgcn_wmma_f32_16x16x32_f16(false, a, false, b, (short)0, c,
                                                false, false);
}

// One wave computes a 16(M) x 64(N) tile; A row-major [M][K], Bt row-major [N][K].
__device__ __forceinline__ void gemm_tile_16x64(const half_t* __restrict__ A,
                                                const half_t* __restrict__ Bt,
                                                int Kd, int m0, int n0, int lane,
                                                v8f acc[4]) {
  for (int k0 = 0; k0 < Kd; k0 += 32) {
    const half_t* arow = A + (size_t)m0 * Kd + k0;
    v16h a = load_frag_a(arow, Kd, lane);
    __builtin_prefetch(arow + 32, 0, 1);   // global_prefetch_b8 (next k-slab)
#pragma unroll
    for (int t = 0; t < 4; ++t) {
      v16h b = load_frag_b(Bt + (size_t)(n0 + 16 * t) * Kd + k0, Kd, lane);
      acc[t] = wmma16(a, b, acc[t]);
    }
  }
}

// ----- async global -> LDS staging (gfx1250, ASYNCcnt) ----------------------
// One 32x32 f16 K tile = 2048 contiguous bytes = 128 threads x 16B
// (one global_load_async_to_lds_b128 issued per wave; 4 waves cover the tile).
__device__ __forceinline__ void issue_async_k(const half_t* gk, half_t* lk, int tid) {
  const unsigned laddr = (unsigned)(unsigned long long)lk + tid * 16;
  const char* g = (const char*)gk + tid * 16;
  asm volatile("global_load_async_to_lds_b128 %0, %1, off"
               :: "v"(laddr), "v"(g) : "memory");
}
// One 32x32 f16 V tile from Vt[d][key]: 32 rows x 64B, row stride NN*2 bytes.
__device__ __forceinline__ void issue_async_v(const half_t* gv, half_t* lv, int tid) {
  const int d = tid >> 2;             // 0..31
  const int cb = (tid & 3) * 16;      // byte offset in row
  const unsigned laddr = (unsigned)(unsigned long long)lv + d * 64 + cb;
  const char* g = (const char*)(gv + (size_t)d * NN) + cb;
  asm volatile("global_load_async_to_lds_b128 %0, %1, off"
               :: "v"(laddr), "v"(g) : "memory");
}

// ----- weight transpose + cast to f16: w[K][N] f32 -> wt[N][K] f16 ----------
__global__ void transpose_cast_kernel(const float* __restrict__ w,
                                      half_t* __restrict__ wt, int K, int N) {
  int idx = blockIdx.x * blockDim.x + threadIdx.x;
  if (idx < K * N) {
    int k = idx / N, n = idx - k * N;
    wt[(size_t)n * K + k] = (half_t)w[idx];
  }
}

// ----- LayerNorm (DIM=256) + cast to f16; one wave per row ------------------
__global__ void ln_cast_kernel(const float* __restrict__ x,
                               const float* __restrict__ g,
                               const float* __restrict__ bta,
                               half_t* __restrict__ out) {
  const int row = blockIdx.x;
  const int lane = threadIdx.x;                 // 32 threads
  const float* xr = x + (size_t)row * DIMX;
  v4f a = *(const v4f*)(xr + lane * 8);
  v4f b = *(const v4f*)(xr + lane * 8 + 4);
  float s = 0.f, sq = 0.f;
#pragma unroll
  for (int i = 0; i < 4; ++i) {
    s += a[i] + b[i];
    sq += a[i] * a[i] + b[i] * b[i];
  }
#pragma unroll
  for (int off = 16; off >= 1; off >>= 1) {
    s  += __shfl_xor(s,  off, 32);
    sq += __shfl_xor(sq, off, 32);
  }
  const float mu = s * (1.0f / DIMX);
  const float var = sq * (1.0f / DIMX) - mu * mu;
  const float rstd = rsqrtf(var + EPSV);
  half_t* orow = out + (size_t)row * DIMX;
#pragma unroll
  for (int i = 0; i < 4; ++i) {
    int c0 = lane * 8 + i, c1 = lane * 8 + 4 + i;
    orow[c0] = (half_t)((a[i] - mu) * rstd * g[c0] + bta[c0]);
    orow[c1] = (half_t)((b[i] - mu) * rstd * g[c1] + bta[c1]);
  }
}

// ----- QKV GEMM: [4096,256] x [256,768]; scatter to Q/K [B,H,N,32], Vt [B,H,32,N]
__global__ void qkv_gemm_kernel(const half_t* __restrict__ A,
                                const half_t* __restrict__ Bt,
                                const float* __restrict__ bias,
                                half_t* __restrict__ qb, half_t* __restrict__ kb,
                                half_t* __restrict__ vtb) {
  const int lane = threadIdx.x & 31;
  const int gw = blockIdx.x * 8 + (threadIdx.x >> 5);
  const int NT = (3 * DIMX) / 64;               // 12
  const int m0 = (gw / NT) * 16, n0 = (gw % NT) * 64;
  v8f acc[4] = {};
  gemm_tile_16x64(A, Bt, DIMX, m0, n0, lane, acc);
  const int hf = lane >> 4, nl = lane & 15;
#pragma unroll
  for (int t = 0; t < 4; ++t) {
    const int col = n0 + t * 16 + nl;
    const float bv = bias[col];
    const int grp = col >> 8;                   // 0=q 1=k 2=v
    const int cc = col & 255;
    const int hd = cc >> 5, d = cc & 31;
#pragma unroll
    for (int i = 0; i < 8; ++i) {
      const int r = m0 + i + 8 * hf;
      const int bi = r >> 11, n = r & (NN - 1);
      const half_t hv = (half_t)(acc[t][i] + bv);
      const size_t bh = (size_t)(bi * HEADS + hd);
      if (grp == 0)      qb[(bh * NN + n) * HD + d] = hv;
      else if (grp == 1) kb[(bh * NN + n) * HD + d] = hv;
      else               vtb[(bh * HD + d) * NN + n] = hv;   // V transposed
    }
  }
}

// ----- flash attention: 4 waves / block share async-staged K/V tiles --------
__global__ void attn_kernel(const half_t* __restrict__ Q,
                            const half_t* __restrict__ Kb,
                            const half_t* __restrict__ Vt,
                            const int* __restrict__ etypes,
                            const unsigned char* __restrict__ nmask,
                            const float* __restrict__ table,
                            half_t* __restrict__ out) {
  __shared__ __align__(16) half_t Kst[2][32 * 32];   // [stage][key][d]
  __shared__ __align__(16) half_t Vst[2][32 * 32];   // [stage][d][key]
  __shared__ __align__(16) half_t P[4][16 * 32];     // per-wave prob relayout
  const int tid = threadIdx.x;                       // 128 threads
  const int wave = tid >> 5, lane = tid & 31;
  const int QG = NN / 64;                            // 32 q-tile groups / head
  const int qg = blockIdx.x % QG;
  const int h  = (blockIdx.x / QG) % HEADS;
  const int b  = blockIdx.x / (QG * HEADS);
  const int q0 = (qg * 4 + wave) * 16;
  const size_t bh = (size_t)(b * HEADS + h);
  const half_t* Qp = Q  + bh * NN * HD;
  const half_t* Kp = Kb + bh * NN * HD;
  const half_t* Vp = Vt + bh * HD * NN;
  const int hf = lane >> 4, nl = lane & 15;
  half_t* Pw = &P[wave][0];

  v16h aQ = load_frag_a(Qp + (size_t)q0 * HD, HD, lane);
  float mrow[8], lrow[8];
  unsigned char nmq[8];
  v8f acc0 = {}, acc1 = {};
#pragma unroll
  for (int i = 0; i < 8; ++i) {
    mrow[i] = -3.0e38f; lrow[i] = 0.f;
    nmq[i] = nmask[b * NN + q0 + i + 8 * hf];
  }

  // preload stage 0 (2 async ops per wave: K slice + V slice)
  issue_async_k(Kp, &Kst[0][0], tid);
  issue_async_v(Vp, &Vst[0][0], tid);

  for (int k0 = 0; k0 < NN; k0 += 32) {
    const int s = (k0 >> 5) & 1;
    if (k0 + 32 < NN) {
      // prefetch next stage into the other buffer, keep 2 ops in flight
      issue_async_k(Kp + (size_t)(k0 + 32) * HD, &Kst[s ^ 1][0], tid);
      issue_async_v(Vp + (k0 + 32), &Vst[s ^ 1][0], tid);
      asm volatile("s_wait_asynccnt 2" ::: "memory");   // stage s landed
    } else {
      asm volatile("s_wait_asynccnt 0" ::: "memory");
    }
    __syncthreads();   // all waves' slices of stage s are in LDS

    v16h bK0 = load_frag_b(&Kst[s][0], HD, lane);
    v16h bK1 = load_frag_b(&Kst[s][16 * HD], HD, lane);
    v8f z = {};
    v8f s0 = wmma16(aQ, bK0, z);
    v8f s1 = wmma16(aQ, bK1, z);
    const int n0c = k0 + nl, n1c = n0c + 16;
    const bool nm0 = nmask[b * NN + n0c] != 0;
    const bool nm1 = nmask[b * NN + n1c] != 0;
#pragma unroll
    for (int i = 0; i < 8; ++i) {
      const int r = q0 + i + 8 * hf;
      const int* ep = etypes + ((size_t)b * NN + r) * NN;
      const int e0 = ep[n0c], e1 = ep[n1c];
      float v0 = s0[i] * SCALEV + table[e0 * HEADS + h];
      float v1 = s1[i] * SCALEV + table[e1 * HEADS + h];
      const bool ok0 = ((e0 != 0) || (r == n0c)) && nmq[i] && nm0;
      const bool ok1 = ((e1 != 0) || (r == n1c)) && nmq[i] && nm1;
      v0 = ok0 ? v0 : -1e9f;
      v1 = ok1 ? v1 : -1e9f;
      // row max / sum across the 16 lanes of this half-wave
      float lmax = fmaxf(v0, v1);
#pragma unroll
      for (int off = 8; off >= 1; off >>= 1)
        lmax = fmaxf(lmax, __shfl_xor(lmax, off, 32));
      const float mnew = fmaxf(mrow[i], lmax);
      const float sf = __expf(mrow[i] - mnew);
      const float p0 = __expf(v0 - mnew);
      const float p1 = __expf(v1 - mnew);
      float ps = p0 + p1;
#pragma unroll
      for (int off = 8; off >= 1; off >>= 1)
        ps += __shfl_xor(ps, off, 32);
      lrow[i] = lrow[i] * sf + ps;
      mrow[i] = mnew;
      acc0[i] *= sf;
      acc1[i] *= sf;
      const int rl = i + 8 * hf;
      Pw[rl * 32 + nl]      = (half_t)p0;   // C-layout -> LDS row-major
      Pw[rl * 32 + 16 + nl] = (half_t)p1;
    }
    // reload P in A-fragment layout (per-wave private region; DS is in-order
    // within a wave), multiply by V (d split into two 16-col halves)
    {
      const half_t* pp = &Pw[nl * 32 + 8 * hf];
      v8h lo = *(const v8h*)pp;
      v8h hi = *(const v8h*)(pp + 16);
      v16h aP = __builtin_shufflevector(lo, hi,
                  0,1,2,3,4,5,6,7,8,9,10,11,12,13,14,15);
      v16h bV0 = load_frag_b(&Vst[s][0], 32, lane);
      v16h bV1 = load_frag_b(&Vst[s][16 * 32], 32, lane);
      acc0 = wmma16(aP, bV0, acc0);
      acc1 = wmma16(aP, bV1, acc1);
    }
    __syncthreads();   // everyone done reading stage s before it is refilled
  }
  // normalize and write [B,N,DIM] at column block h*32
#pragma unroll
  for (int i = 0; i < 8; ++i) {
    const int r = q0 + i + 8 * hf;
    const float inv = 1.f / lrow[i];
    const size_t base = ((size_t)b * NN + r) * DIMX + h * HD;
    out[base + nl]      = (half_t)(acc0[i] * inv);
    out[base + 16 + nl] = (half_t)(acc1[i] * inv);
  }
}

// ----- proj GEMM + bias + node mask + residual -> x1 (f32) ------------------
__global__ void proj_gemm_kernel(const half_t* __restrict__ A,
                                 const half_t* __restrict__ Bt,
                                 const float* __restrict__ bias,
                                 const float* __restrict__ xin,
                                 const unsigned char* __restrict__ nmask,
                                 float* __restrict__ x1) {
  const int lane = threadIdx.x & 31;
  const int gw = blockIdx.x * 8 + (threadIdx.x >> 5);
  const int NT = DIMX / 64;                     // 4
  const int m0 = (gw / NT) * 16, n0 = (gw % NT) * 64;
  v8f acc[4] = {};
  gemm_tile_16x64(A, Bt, DIMX, m0, n0, lane, acc);
  const int hf = lane >> 4, nl = lane & 15;
#pragma unroll
  for (int t = 0; t < 4; ++t) {
    const int col = n0 + t * 16 + nl;
    const float bv = bias[col];
#pragma unroll
    for (int i = 0; i < 8; ++i) {
      const int r = m0 + i + 8 * hf;
      const int bi = r >> 11, n = r & (NN - 1);
      const float mv = nmask[bi * NN + n] ? 1.f : 0.f;
      const size_t o = (size_t)r * DIMX + col;
      x1[o] = xin[o] + (acc[t][i] + bv) * mv;
    }
  }
}

// ----- FFN1 GEMM + bias + exact GELU -> f16 ---------------------------------
__global__ void ffn1_gemm_kernel(const half_t* __restrict__ A,
                                 const half_t* __restrict__ Bt,
                                 const float* __restrict__ bias,
                                 half_t* __restrict__ g16) {
  const int lane = threadIdx.x & 31;
  const int gw = blockIdx.x * 8 + (threadIdx.x >> 5);
  const int NT = FF / 64;                       // 16
  const int m0 = (gw / NT) * 16, n0 = (gw % NT) * 64;
  v8f acc[4] = {};
  gemm_tile_16x64(A, Bt, DIMX, m0, n0, lane, acc);
  const int hf = lane >> 4, nl = lane & 15;
#pragma unroll
  for (int t = 0; t < 4; ++t) {
    const int col = n0 + t * 16 + nl;
    const float bv = bias[col];
#pragma unroll
    for (int i = 0; i < 8; ++i) {
      const int r = m0 + i + 8 * hf;
      float v = acc[t][i] + bv;
      v = 0.5f * v * (1.f + erff(v * 0.70710678118654752f));  // exact gelu
      g16[(size_t)r * FF + col] = (half_t)v;
    }
  }
}

// ----- FFN2 GEMM + bias + mask + residual -> d_out (f32) --------------------
__global__ void ffn2_gemm_kernel(const half_t* __restrict__ A,
                                 const half_t* __restrict__ Bt,
                                 const float* __restrict__ bias,
                                 const float* __restrict__ x1,
                                 const unsigned char* __restrict__ nmask,
                                 float* __restrict__ outp) {
  const int lane = threadIdx.x & 31;
  const int gw = blockIdx.x * 8 + (threadIdx.x >> 5);
  const int NT = DIMX / 64;                     // 4
  const int m0 = (gw / NT) * 16, n0 = (gw % NT) * 64;
  v8f acc[4] = {};
  gemm_tile_16x64(A, Bt, FF, m0, n0, lane, acc);
  const int hf = lane >> 4, nl = lane & 15;
#pragma unroll
  for (int t = 0; t < 4; ++t) {
    const int col = n0 + t * 16 + nl;
    const float bv = bias[col];
#pragma unroll
    for (int i = 0; i < 8; ++i) {
      const int r = m0 + i + 8 * hf;
      const int bi = r >> 11, n = r & (NN - 1);
      const float mv = nmask[bi * NN + n] ? 1.f : 0.f;
      const size_t o = (size_t)r * DIMX + col;
      outp[o] = x1[o] + (acc[t][i] + bv) * mv;
    }
  }
}

// ---------------------------------------------------------------------------
extern "C" void kernel_launch(void* const* d_in, const int* in_sizes, int n_in,
                              void* d_out, int out_size, void* d_ws, size_t ws_size,
                              hipStream_t stream) {
  const float* x      = (const float*)d_in[0];
  const int*   etypes = (const int*)d_in[1];
  const unsigned char* nmask = (const unsigned char*)d_in[2];  // numpy bool (1B)
  const float* qkv_w  = (const float*)d_in[3];
  const float* qkv_b  = (const float*)d_in[4];
  const float* proj_w = (const float*)d_in[5];
  const float* proj_b = (const float*)d_in[6];
  const float* table  = (const float*)d_in[7];
  const float* ln1_g  = (const float*)d_in[8];
  const float* ln1_b  = (const float*)d_in[9];
  const float* ln2_g  = (const float*)d_in[10];
  const float* ln2_b  = (const float*)d_in[11];
  const float* ffn_w1 = (const float*)d_in[12];
  const float* ffn_b1 = (const float*)d_in[13];
  const float* ffn_w2 = (const float*)d_in[14];
  const float* ffn_b2 = (const float*)d_in[15];
  float* outp = (float*)d_out;

  // workspace layout (~24.6 MB total)
  char* ws = (char*)d_ws;
  half_t* qkv_wt = (half_t*)ws;                 ws += (size_t)768 * 256 * 2;
  half_t* proj_wt = (half_t*)ws;                ws += (size_t)256 * 256 * 2;
  half_t* w1t = (half_t*)ws;                    ws += (size_t)1024 * 256 * 2;
  half_t* w2t = (half_t*)ws;                    ws += (size_t)256 * 1024 * 2;
  half_t* h16 = (half_t*)ws;                    ws += (size_t)MROWS * DIMX * 2;
  half_t* qb  = (half_t*)ws;                    ws += (size_t)BB * HEADS * NN * HD * 2;
  half_t* kb  = (half_t*)ws;                    ws += (size_t)BB * HEADS * NN * HD * 2;
  half_t* vtb = (half_t*)ws;                    ws += (size_t)BB * HEADS * HD * NN * 2;
  half_t* attn16 = (half_t*)ws;                 ws += (size_t)MROWS * DIMX * 2;
  float*  x1  = (float*)ws;                     ws += (size_t)MROWS * DIMX * 4;
  half_t* g16 = (half_t*)ws;                    ws += (size_t)MROWS * FF * 2;

  // 1) f16 transposed weights
  transpose_cast_kernel<<<(256 * 768 + 255) / 256, 256, 0, stream>>>(qkv_w, qkv_wt, 256, 768);
  transpose_cast_kernel<<<(256 * 256 + 255) / 256, 256, 0, stream>>>(proj_w, proj_wt, 256, 256);
  transpose_cast_kernel<<<(256 * 1024 + 255) / 256, 256, 0, stream>>>(ffn_w1, w1t, 256, 1024);
  transpose_cast_kernel<<<(1024 * 256 + 255) / 256, 256, 0, stream>>>(ffn_w2, w2t, 1024, 256);

  // 2) LN1 + cast
  ln_cast_kernel<<<MROWS, 32, 0, stream>>>(x, ln1_g, ln1_b, h16);

  // 3) QKV GEMM (M=4096,N=768,K=256): 3072 waves
  qkv_gemm_kernel<<<(256 * 12) / 8, 256, 0, stream>>>(h16, qkv_wt, qkv_b, qb, kb, vtb);

  // 4) flash attention: 4 waves/block share async-staged K/V tiles
  attn_kernel<<<BB * HEADS * (NN / 64), 128, 0, stream>>>(qb, kb, vtb, etypes, nmask,
                                                          table, attn16);

  // 5) proj + mask + residual -> x1
  proj_gemm_kernel<<<(256 * 4) / 8, 256, 0, stream>>>(attn16, proj_wt, proj_b, x,
                                                      nmask, x1);

  // 6) LN2 + cast (reuse h16)
  ln_cast_kernel<<<MROWS, 32, 0, stream>>>(x1, ln2_g, ln2_b, h16);

  // 7) FFN1 + gelu (M=4096,N=1024,K=256)
  ffn1_gemm_kernel<<<(256 * 16) / 8, 256, 0, stream>>>(h16, w1t, ffn_b1, g16);

  // 8) FFN2 + mask + residual (M=4096,N=256,K=1024) -> d_out
  ffn2_gemm_kernel<<<(256 * 4) / 8, 256, 0, stream>>>(g16, w2t, ffn_b2, x1, nmask, outp);
}
